// SAPM_70635032150802
// MI455X (gfx1250) — matrix operand order repaired and checked by reading, not compiled, not run on hardware
//
#include <hip/hip_runtime.h>
#include <cstdint>
#include <cstddef>

// ---------------------------------------------------------------------------
// Types
// ---------------------------------------------------------------------------
typedef __bf16 bf16;
typedef __attribute__((ext_vector_type(16))) __bf16 v16bf;
typedef __attribute__((ext_vector_type(8)))  __bf16 v8bf;
typedef __attribute__((ext_vector_type(8)))  float  v8f;
typedef __attribute__((ext_vector_type(4)))  float  v4f;

#define DEV static __device__ __forceinline__

// ---------------------------------------------------------------------------
// Problem constants (from reference setup_inputs)
// ---------------------------------------------------------------------------
constexpr int BN  = 8;
constexpr int CC  = 256;
constexpr int HH  = 100;
constexpr int WW  = 100;
constexpr int HWN = HH * WW;        // 10000
constexpr int HWP = 10016;          // padded to multiple of 32 (WMMA K)
constexpr int QN  = 300;
constexpr int QP  = 304;            // Q padded to multiple of 16 (A rows, BMM)
constexpr int QW  = 320;            // Q padded to multiple of 64 (pw3 weight rows)
constexpr int RQ  = BN * QN;        // 2400 rows in CR stage
constexpr int RQP = 2432;           // padded to multiple of 64 (FC B columns)
constexpr float TAU = 1.2f;

// ---------------------------------------------------------------------------
// Helpers
// ---------------------------------------------------------------------------
DEV float lsq_q(float v, float a) {
  // LSQ forward (grad_scale is identity in fwd): round(clip(v/a, -8, 7)) * a
  float s = v / a;
  s = fminf(fmaxf(s, -8.0f), 7.0f);
  return rintf(s) * a;
}

DEV v16bf mk16(const bf16* p0, const bf16* p1) {
  v8bf lo = *(const v8bf*)p0;
  v8bf hi = *(const v8bf*)p1;
  v16bf r;
#pragma unroll
  for (int i = 0; i < 8; ++i) { r[i] = lo[i]; r[8 + i] = hi[i]; }
  return r;
}

DEV v8f zero8() {
  v8f r;
#pragma unroll
  for (int i = 0; i < 8; ++i) r[i] = 0.0f;
  return r;
}

DEV v8f wmma_bf16(v16bf a, v16bf b, v8f c) {
  return __builtin_amdgcn_wmma_f32_16x16x32_bf16(false, a, false, b, (short)0,
                                                 c, false, false);
}

// ---------------------------------------------------------------------------
// Weight fake-quantization (LSQ, nbits=4)
// ---------------------------------------------------------------------------
__global__ void k_quant_f32(const float* __restrict__ w, const float* __restrict__ pa,
                            float* __restrict__ out, int n) {
  int i = blockIdx.x * 256 + threadIdx.x;
  if (i < n) out[i] = lsq_q(w[i], pa[0]);
}

__global__ void k_quant_bf16(const float* __restrict__ w, const float* __restrict__ pa,
                             bf16* __restrict__ out, int n) {
  int i = blockIdx.x * 256 + threadIdx.x;
  if (i < n) out[i] = (bf16)lsq_q(w[i], pa[0]);
}

// ---------------------------------------------------------------------------
// Input prep: x NCHW f32 -> xb NHWC bf16 (dw1 input) and Fq NCHW-padded bf16
// (activation-quantized x for the weighted-pooling BMM B-matrix).
// ---------------------------------------------------------------------------
__global__ void k_prep(const float* __restrict__ x, const float* __restrict__ pAF,
                       bf16* __restrict__ xb, bf16* __restrict__ fq) {
  int i = blockIdx.x * 256 + threadIdx.x;           // over B*C*HW, NCHW order
  if (i >= BN * CC * HWN) return;
  int s = i % HWN;
  int c = (i / HWN) % CC;
  int b = i / (HWN * CC);
  float v = x[i];
  xb[((size_t)b * HWN + s) * CC + c] = (bf16)v;
  fq[((size_t)b * CC + c) * HWP + s] = (bf16)lsq_q(v, pAF[0]);
}

__global__ void k_padzero(bf16* __restrict__ fq) {
  int i = blockIdx.x * 256 + threadIdx.x;           // over B*C*(HWP-HW)
  int pad = HWP - HWN;
  if (i >= BN * CC * pad) return;
  int rc = i / pad;
  int s  = HWN + (i % pad);
  fq[(size_t)rc * HWP + s] = (bf16)0.0f;
}

// ---------------------------------------------------------------------------
// Depthwise conv (NHWC bf16 -> NHWC bf16), quantized f32 weights [C, KS*KS]
// ---------------------------------------------------------------------------
template <int KS, int PAD>
__global__ void k_dw(const bf16* __restrict__ in, const float* __restrict__ wq,
                     const float* __restrict__ bias, bf16* __restrict__ out) {
  int i = blockIdx.x * 256 + threadIdx.x;           // over B*HW*C, NHWC order
  if (i >= BN * HWN * CC) return;
  int c  = i % CC;
  int sp = (i / CC) % HWN;
  int b  = i / (CC * HWN);
  int y0 = sp / WW, x0 = sp % WW;
  const bf16* inb = in + (size_t)b * HWN * CC;
  const float* wc = wq + c * KS * KS;
  float acc = bias[c];
#pragma unroll
  for (int dy = 0; dy < KS; ++dy) {
    int yy = y0 + dy - PAD;
    if (yy < 0 || yy >= HH) continue;
#pragma unroll
    for (int dx = 0; dx < KS; ++dx) {
      int xx = x0 + dx - PAD;
      if (xx < 0 || xx >= WW) continue;
      acc += (float)inb[(size_t)(yy * WW + xx) * CC + c] * wc[dy * KS + dx];
    }
  }
  out[i] = (bf16)acc;
}

// ---------------------------------------------------------------------------
// Pointwise conv as WMMA GEMM (software-pipelined):
//   y[b, co, s] = scale * (bias[co] + sum_ci act[b, s, ci] * w[co, ci])
// act: NHWC bf16 [B, HW, C]; w: bf16 [CoutPad, C] (rows padded to tile mult);
// out: NCHW f32 [B, Cout, HW].
// Wave tile: 16 (s) x 64 (co). A = act tile (M=s,K=ci), B = w^T tile.
// Fragments for step k+32 are loaded before the WMMAs of step k issue, so
// the loads overlap with matrix math instead of stalling on s_wait_loadcnt 0.
// ---------------------------------------------------------------------------
__global__ void __launch_bounds__(32)
k_pw(const bf16* __restrict__ act, const bf16* __restrict__ w,
     const float* __restrict__ bias, float* __restrict__ out,
     int Cout, float scale) {
  const int lane = threadIdx.x;
  const int l15  = lane & 15;
  const int kbA  = (lane < 16) ? 0 : 16;
  const int kbB  = (lane < 16) ? 0 : 16;
  const int b    = blockIdx.z;
  const int m0   = blockIdx.y * 16;                 // s tile
  const int n00  = blockIdx.x * 64;                 // co group (4 tiles)

  const bf16* Ab = act + ((size_t)b * HWN + (m0 + l15)) * CC + kbA;
  const bf16* B0 = w + (size_t)(n00 + l15) * CC + kbB;
  const bf16* B1 = w + (size_t)(n00 + 16 + l15) * CC + kbB;
  const bf16* B2 = w + (size_t)(n00 + 32 + l15) * CC + kbB;
  const bf16* B3 = w + (size_t)(n00 + 48 + l15) * CC + kbB;

  v8f acc0 = zero8(), acc1 = zero8(), acc2 = zero8(), acc3 = zero8();

  // prologue: stage 0 fragments
  v16bf aC  = mk16(Ab, Ab + 8);
  v16bf b0C = mk16(B0, B0 + 8);
  v16bf b1C = mk16(B1, B1 + 8);
  v16bf b2C = mk16(B2, B2 + 8);
  v16bf b3C = mk16(B3, B3 + 8);

#pragma unroll
  for (int k0 = 32; k0 < CC; k0 += 32) {
    v16bf aN  = mk16(Ab + k0, Ab + k0 + 8);
    v16bf b0N = mk16(B0 + k0, B0 + k0 + 8);
    v16bf b1N = mk16(B1 + k0, B1 + k0 + 8);
    v16bf b2N = mk16(B2 + k0, B2 + k0 + 8);
    v16bf b3N = mk16(B3 + k0, B3 + k0 + 8);
    acc0 = wmma_bf16(aC, b0C, acc0);
    acc1 = wmma_bf16(aC, b1C, acc1);
    acc2 = wmma_bf16(aC, b2C, acc2);
    acc3 = wmma_bf16(aC, b3C, acc3);
    aC = aN; b0C = b0N; b1C = b1N; b2C = b2N; b3C = b3N;
  }
  acc0 = wmma_bf16(aC, b0C, acc0);
  acc1 = wmma_bf16(aC, b1C, acc1);
  acc2 = wmma_bf16(aC, b2C, acc2);
  acc3 = wmma_bf16(aC, b3C, acc3);

  const int sb = m0 + ((lane < 16) ? 0 : 8);        // 8 consecutive s values
  v8f accs[4] = {acc0, acc1, acc2, acc3};
#pragma unroll
  for (int t = 0; t < 4; ++t) {
    int co = n00 + t * 16 + l15;
    if (co >= Cout) continue;
    float bb = bias[co];
    float* po = out + ((size_t)b * Cout + co) * HWN + sb;
    v4f r0, r1;
#pragma unroll
    for (int i = 0; i < 4; ++i) {
      r0[i] = (accs[t][i] + bb) * scale;
      r1[i] = (accs[t][4 + i] + bb) * scale;
    }
    *(v4f*)po = r0;
    *(v4f*)(po + 4) = r1;
  }
}

// ---------------------------------------------------------------------------
// GroupNorm (32 groups) over NCHW f32: stats then apply+PReLU -> NHWC bf16
// ---------------------------------------------------------------------------
__global__ void k_gnstats(const float* __restrict__ y, float* __restrict__ stats) {
  __shared__ float s1[256];
  __shared__ float s2[256];
  const int bg  = blockIdx.x;                       // b*32 + g
  const int tid = threadIdx.x;
  const int b = bg >> 5, g = bg & 31;
  const float* yb = y + ((size_t)b * CC + g * 8) * HWN;   // 80000 contiguous
  const int n = 8 * HWN;
  float sum = 0.f, ss = 0.f;
  for (int i = tid; i < n; i += 256) { float v = yb[i]; sum += v; ss += v * v; }
  s1[tid] = sum; s2[tid] = ss;
  __syncthreads();
  for (int o = 128; o > 0; o >>= 1) {
    if (tid < o) { s1[tid] += s1[tid + o]; s2[tid] += s2[tid + o]; }
    __syncthreads();
  }
  if (tid == 0) {
    float mean = s1[0] / (float)n;
    float var  = s2[0] / (float)n - mean * mean;
    stats[bg * 2 + 0] = mean;
    stats[bg * 2 + 1] = rsqrtf(var + 1e-5f);
  }
}

__global__ void k_gnapply(const float* __restrict__ y, const float* __restrict__ stats,
                          const float* __restrict__ gw, const float* __restrict__ gb,
                          const float* __restrict__ pamp, bf16* __restrict__ h) {
  int i = blockIdx.x * 256 + threadIdx.x;           // NCHW order (coalesced read)
  if (i >= BN * CC * HWN) return;
  int s = i % HWN;
  int c = (i / HWN) % CC;
  int b = i / (HWN * CC);
  int g = c >> 3;
  float m = stats[(b * 32 + g) * 2 + 0];
  float r = stats[(b * 32 + g) * 2 + 1];
  float v = (y[i] - m) * r * gw[c] + gb[c];
  float a = pamp[0];
  v = (v >= 0.f) ? v : a * v;
  h[((size_t)b * HWN + s) * CC + c] = (bf16)v;
}

// ---------------------------------------------------------------------------
// Spatial softmax over HW per (b,q) row of a[B,Q,HW], epilogue LSQ->bf16,
// write padded Aq[B,QP,HWP] (spatial pad = 0; q pad rows left untouched).
// ---------------------------------------------------------------------------
__global__ void k_softmax(const float* __restrict__ a, const float* __restrict__ pA,
                          bf16* __restrict__ aq) {
  __shared__ float red[256];
  const int row = blockIdx.x;                       // b*QN + q
  const int b = row / QN, q = row % QN;
  const int tid = threadIdx.x;
  const float* ar = a + (size_t)row * HWN;
  float mx = -3.0e38f;
  for (int s = tid; s < HWN; s += 256) mx = fmaxf(mx, ar[s]);
  red[tid] = mx;
  __syncthreads();
  for (int o = 128; o > 0; o >>= 1) {
    if (tid < o) red[tid] = fmaxf(red[tid], red[tid + o]);
    __syncthreads();
  }
  mx = red[0];
  __syncthreads();
  float sum = 0.f;
  for (int s = tid; s < HWN; s += 256) sum += __expf(ar[s] - mx);
  red[tid] = sum;
  __syncthreads();
  for (int o = 128; o > 0; o >>= 1) {
    if (tid < o) red[tid] += red[tid + o];
    __syncthreads();
  }
  float inv = 1.0f / red[0];
  float al  = pA[0];
  bf16* qr = aq + ((size_t)b * QP + q) * HWP;
  for (int s = tid; s < HWP; s += 256) {
    float o = 0.f;
    if (s < HWN) o = lsq_q(__expf(ar[s] - mx) * inv, al);
    qr[s] = (bf16)o;
  }
}

// ---------------------------------------------------------------------------
// Weighted pooling BMM (software-pipelined): F_P[b,q,c] = sum_s Aq*Fq
// A = Aq tile (M=q, K=s) from [B,QP,HWP]; B = Fq tile (K=s, N=c) from
// [B,C,HWP]. Wave tile 16q x 64c. Branch-free rolled K loop (313 steps) with
// next-step fragments loaded before current-step WMMAs issue.
// ---------------------------------------------------------------------------
__global__ void __launch_bounds__(32)
k_bmm(const bf16* __restrict__ aq, const bf16* __restrict__ fq,
      float* __restrict__ fp, bf16* __restrict__ fpb) {
  const int lane = threadIdx.x;
  const int l15  = lane & 15;
  const int kbA  = (lane < 16) ? 0 : 16;
  const int kbB  = (lane < 16) ? 0 : 16;
  const int b    = blockIdx.z;
  const int m0   = blockIdx.y * 16;                 // q tile
  const int n00  = blockIdx.x * 64;                 // c group

  const bf16* Arow = aq + ((size_t)b * QP + m0 + l15) * HWP + kbA;
  const bf16* Fb   = fq + (size_t)b * CC * HWP + kbB;
  const bf16* B0 = Fb + (size_t)(n00 + l15) * HWP;
  const bf16* B1 = Fb + (size_t)(n00 + 16 + l15) * HWP;
  const bf16* B2 = Fb + (size_t)(n00 + 32 + l15) * HWP;
  const bf16* B3 = Fb + (size_t)(n00 + 48 + l15) * HWP;

  v8f acc0 = zero8(), acc1 = zero8(), acc2 = zero8(), acc3 = zero8();

  v16bf aC  = mk16(Arow, Arow + 8);
  v16bf b0C = mk16(B0, B0 + 8);
  v16bf b1C = mk16(B1, B1 + 8);
  v16bf b2C = mk16(B2, B2 + 8);
  v16bf b3C = mk16(B3, B3 + 8);

  for (int k0 = 32; k0 < HWP; k0 += 32) {
    __builtin_prefetch((const void*)(Arow + k0 + 32), 0, 1);
    v16bf aN  = mk16(Arow + k0, Arow + k0 + 8);
    v16bf b0N = mk16(B0 + k0, B0 + k0 + 8);
    v16bf b1N = mk16(B1 + k0, B1 + k0 + 8);
    v16bf b2N = mk16(B2 + k0, B2 + k0 + 8);
    v16bf b3N = mk16(B3 + k0, B3 + k0 + 8);
    acc0 = wmma_bf16(aC, b0C, acc0);
    acc1 = wmma_bf16(aC, b1C, acc1);
    acc2 = wmma_bf16(aC, b2C, acc2);
    acc3 = wmma_bf16(aC, b3C, acc3);
    aC = aN; b0C = b0N; b1C = b1N; b2C = b2N; b3C = b3N;
  }
  acc0 = wmma_bf16(aC, b0C, acc0);
  acc1 = wmma_bf16(aC, b1C, acc1);
  acc2 = wmma_bf16(aC, b2C, acc2);
  acc3 = wmma_bf16(aC, b3C, acc3);

  const int qb = m0 + ((lane < 16) ? 0 : 8);
  v8f accs[4] = {acc0, acc1, acc2, acc3};
#pragma unroll
  for (int t = 0; t < 4; ++t) {
    int c = n00 + t * 16 + l15;
#pragma unroll
    for (int r = 0; r < 8; ++r) {
      int q = qb + r;
      if (q >= QN) continue;
      size_t idx = ((size_t)b * QN + q) * CC + c;
      fp[idx]  = accs[t][r];
      fpb[idx] = (bf16)accs[t][r];
    }
  }
}

// ---------------------------------------------------------------------------
// Shared FC GEMM core (software-pipelined): D[co,row] = sum_c W[co,c]*In[row,c]
// A = Wq (M=co,K=c), B = In^T (K=c,N=row), rows padded to RQP (garbage pad
// columns discarded by store guards). Wave tile 16co x 64rows.
// ---------------------------------------------------------------------------
DEV void fc_core(const bf16* __restrict__ wq, const bf16* __restrict__ inb,
                 int m0, int n00, int lane, v8f accs[4]) {
  const int l15 = lane & 15;
  const int kbA = (lane < 16) ? 0 : 16;
  const int kbB = (lane < 16) ? 0 : 16;
  const bf16* Aw = wq + (size_t)(m0 + l15) * CC + kbA;
  const bf16* B0 = inb + (size_t)(n00 + l15) * CC + kbB;
  const bf16* B1 = inb + (size_t)(n00 + 16 + l15) * CC + kbB;
  const bf16* B2 = inb + (size_t)(n00 + 32 + l15) * CC + kbB;
  const bf16* B3 = inb + (size_t)(n00 + 48 + l15) * CC + kbB;

  v8f a0 = zero8(), a1 = zero8(), a2 = zero8(), a3 = zero8();

  v16bf aC  = mk16(Aw, Aw + 8);
  v16bf b0C = mk16(B0, B0 + 8);
  v16bf b1C = mk16(B1, B1 + 8);
  v16bf b2C = mk16(B2, B2 + 8);
  v16bf b3C = mk16(B3, B3 + 8);

#pragma unroll
  for (int k0 = 32; k0 < CC; k0 += 32) {
    v16bf aN  = mk16(Aw + k0, Aw + k0 + 8);
    v16bf b0N = mk16(B0 + k0, B0 + k0 + 8);
    v16bf b1N = mk16(B1 + k0, B1 + k0 + 8);
    v16bf b2N = mk16(B2 + k0, B2 + k0 + 8);
    v16bf b3N = mk16(B3 + k0, B3 + k0 + 8);
    a0 = wmma_bf16(aC, b0C, a0);
    a1 = wmma_bf16(aC, b1C, a1);
    a2 = wmma_bf16(aC, b2C, a2);
    a3 = wmma_bf16(aC, b3C, a3);
    aC = aN; b0C = b0N; b1C = b1N; b2C = b2N; b3C = b3N;
  }
  a0 = wmma_bf16(aC, b0C, a0);
  a1 = wmma_bf16(aC, b1C, a1);
  a2 = wmma_bf16(aC, b2C, a2);
  a3 = wmma_bf16(aC, b3C, a3);

  accs[0] = a0; accs[1] = a1; accs[2] = a2; accs[3] = a3;
}

// FC1: hcr = PReLU(F_P @ W1q^T + b1) -> bf16 [RQP, 256]
__global__ void __launch_bounds__(32)
k_fc1(const bf16* __restrict__ wq, const bf16* __restrict__ inb,
      const float* __restrict__ bias, const float* __restrict__ pcr,
      bf16* __restrict__ outb) {
  const int lane = threadIdx.x;
  const int l15  = lane & 15;
  const int m0   = blockIdx.y * 16;                 // co tile
  const int n00  = blockIdx.x * 64;                 // row group
  v8f accs[4];
  fc_core(wq, inb, m0, n00, lane, accs);

  const int cob = m0 + ((lane < 16) ? 0 : 8);
  const float am = pcr[0];
#pragma unroll
  for (int t = 0; t < 4; ++t) {
    int row = n00 + t * 16 + l15;
    if (row >= RQ) continue;
    v8bf o;
#pragma unroll
    for (int i = 0; i < 8; ++i) {
      float v = accs[t][i] + bias[cob + i];
      v = (v >= 0.f) ? v : am * v;
      o[i] = (bf16)v;
    }
    *(v8bf*)(outb + (size_t)row * CC + cob) = o;
  }
}

// FC2 + sigmoid + gate: out = lsq(sigmoid(hcr @ W2q^T + b2), aFC) * lsq(F_P, aFP)
__global__ void __launch_bounds__(32)
k_fc2(const bf16* __restrict__ wq, const bf16* __restrict__ inb,
      const float* __restrict__ bias, const float* __restrict__ fp,
      const float* __restrict__ pAFC, const float* __restrict__ pAFP,
      float* __restrict__ out) {
  const int lane = threadIdx.x;
  const int l15  = lane & 15;
  const int m0   = blockIdx.y * 16;
  const int n00  = blockIdx.x * 64;
  v8f accs[4];
  fc_core(wq, inb, m0, n00, lane, accs);

  const int cob = m0 + ((lane < 16) ? 0 : 8);
  const float aFC = pAFC[0];
  const float aFP = pAFP[0];
#pragma unroll
  for (int t = 0; t < 4; ++t) {
    int row = n00 + t * 16 + l15;
    if (row >= RQ) continue;
    const float* fpr = fp + (size_t)row * CC + cob;
    v4f r0, r1;
#pragma unroll
    for (int i = 0; i < 8; ++i) {
      float v = accs[t][i] + bias[cob + i];
      v = 1.0f / (1.0f + __expf(-v));               // sigmoid
      float g = lsq_q(v, aFC) * lsq_q(fpr[i], aFP);
      if (i < 4) r0[i] = g; else r1[i - 4] = g;
    }
    float* po = out + (size_t)row * CC + cob;
    *(v4f*)po = r0;
    *(v4f*)(po + 4) = r1;
  }
}

// ---------------------------------------------------------------------------
// Host-side orchestration
// ---------------------------------------------------------------------------
extern "C" void kernel_launch(void* const* d_in, const int* in_sizes, int n_in,
                              void* d_out, int out_size, void* d_ws, size_t ws_size,
                              hipStream_t stream) {
  (void)in_sizes; (void)n_in; (void)out_size; (void)ws_size;

  const float* x       = (const float*)d_in[0];
  const float* c1dw_w  = (const float*)d_in[1];
  const float* c1dw_b  = (const float*)d_in[2];
  const float* c1dw_a  = (const float*)d_in[3];
  const float* c1pw_w  = (const float*)d_in[4];
  const float* c1pw_b  = (const float*)d_in[5];
  const float* c1pw_a  = (const float*)d_in[6];
  const float* c2dw_w  = (const float*)d_in[7];
  const float* c2dw_b  = (const float*)d_in[8];
  const float* c2dw_a  = (const float*)d_in[9];
  const float* c2pw_w  = (const float*)d_in[10];
  const float* c2pw_b  = (const float*)d_in[11];
  const float* c2pw_a  = (const float*)d_in[12];
  const float* c3dw_w  = (const float*)d_in[13];
  const float* c3dw_b  = (const float*)d_in[14];
  const float* c3dw_a  = (const float*)d_in[15];
  const float* c3pw_w  = (const float*)d_in[16];
  const float* c3pw_b  = (const float*)d_in[17];
  const float* c3pw_a  = (const float*)d_in[18];
  const float* gn1_w   = (const float*)d_in[19];
  const float* gn1_b   = (const float*)d_in[20];
  const float* gn2_w   = (const float*)d_in[21];
  const float* gn2_b   = (const float*)d_in[22];
  const float* pr_amp  = (const float*)d_in[23];
  const float* fc1_w   = (const float*)d_in[24];
  const float* fc1_b   = (const float*)d_in[25];
  const float* fc1_a   = (const float*)d_in[26];
  const float* fc2_w   = (const float*)d_in[27];
  const float* fc2_b   = (const float*)d_in[28];
  const float* fc2_a   = (const float*)d_in[29];
  const float* pr_cr   = (const float*)d_in[30];
  const float* aA_a    = (const float*)d_in[31];
  const float* aF_a    = (const float*)d_in[32];
  const float* aFC_a   = (const float*)d_in[33];
  const float* aFP_a   = (const float*)d_in[34];

  // -------- workspace layout (lifetime-based region reuse) --------
  char* ws = (char*)d_ws;
  size_t off = 0;
  auto alloc = [&](size_t sz) -> size_t {
    size_t o = off;
    off = (off + sz + 255) & ~(size_t)255;
    return o;
  };
  const size_t SZ_NHWC = (size_t)BN * HWN * CC * sizeof(bf16);   // 40.96 MB
  const size_t SZ_FQ   = (size_t)BN * CC * HWP * sizeof(bf16);   // 41.03 MB
  const size_t SZ_Y    = (size_t)BN * CC * HWN * sizeof(float);  // 81.92 MB
  const size_t SZ_A    = (size_t)BN * QN * HWN * sizeof(float);  // 96.00 MB
  const size_t SZ_AQ   = (size_t)BN * QP * HWP * sizeof(bf16);   // 48.72 MB <= SZ_Y

  size_t oR1 = alloc(SZ_NHWC);   // xb -> d2 -> d3
  size_t oR2 = alloc(SZ_NHWC);   // d1 -> h2
  size_t oR3 = alloc(SZ_Y > SZ_AQ ? SZ_Y : SZ_AQ);  // y1 -> y2 -> Aq
  size_t oR4 = alloc(SZ_NHWC);   // h1
  size_t oR5 = alloc(SZ_A);      // a (pre-softmax logits)
  size_t oFQ = alloc(SZ_FQ);     // Fq (quantized x, NCHW padded)

  size_t oFP   = alloc((size_t)RQ * CC * sizeof(float));
  size_t oFPB  = alloc((size_t)RQP * CC * sizeof(bf16));  // row-padded
  size_t oHCR  = alloc((size_t)RQP * CC * sizeof(bf16));  // row-padded
  size_t oQD1  = alloc((size_t)CC * 25 * sizeof(float));
  size_t oQD2  = alloc((size_t)CC * 9 * sizeof(float));
  size_t oQD3  = alloc((size_t)CC * 9 * sizeof(float));
  size_t oQP1  = alloc((size_t)CC * CC * sizeof(bf16));
  size_t oQP2  = alloc((size_t)CC * CC * sizeof(bf16));
  size_t oQP3  = alloc((size_t)QW * CC * sizeof(bf16));   // row-padded
  size_t oQF1  = alloc((size_t)CC * CC * sizeof(bf16));
  size_t oQF2  = alloc((size_t)CC * CC * sizeof(bf16));
  size_t oST   = alloc((size_t)BN * 32 * 2 * sizeof(float));

  bf16*  xb   = (bf16*)(ws + oR1);   // then d2, d3 (same size/layout)
  bf16*  d1   = (bf16*)(ws + oR2);   // then h2
  float* y    = (float*)(ws + oR3);  // y1, y2
  bf16*  aqb  = (bf16*)(ws + oR3);   // Aq (after y2 dead)
  bf16*  h1   = (bf16*)(ws + oR4);
  float* amap = (float*)(ws + oR5);
  bf16*  fq   = (bf16*)(ws + oFQ);
  float* fpv  = (float*)(ws + oFP);
  bf16*  fpb  = (bf16*)(ws + oFPB);
  bf16*  hcrb = (bf16*)(ws + oHCR);
  float* qd1  = (float*)(ws + oQD1);
  float* qd2  = (float*)(ws + oQD2);
  float* qd3  = (float*)(ws + oQD3);
  bf16*  qp1  = (bf16*)(ws + oQP1);
  bf16*  qp2  = (bf16*)(ws + oQP2);
  bf16*  qp3  = (bf16*)(ws + oQP3);
  bf16*  qf1  = (bf16*)(ws + oQF1);
  bf16*  qf2  = (bf16*)(ws + oQF2);
  float* st   = (float*)(ws + oST);

  bf16* d2 = xb;   // region R1 reuse
  bf16* d3 = xb;
  bf16* h2 = d1;   // region R2 reuse

  float* outp = (float*)d_out;

  auto blks = [](long long n) { return (unsigned)((n + 255) / 256); };
  const long long NX = (long long)BN * CC * HWN;    // 20.48M elements

  // ---- 1) quantize weights ----
  k_quant_f32 <<<blks(CC * 25), 256, 0, stream>>>(c1dw_w, c1dw_a, qd1, CC * 25);
  k_quant_f32 <<<blks(CC * 9),  256, 0, stream>>>(c2dw_w, c2dw_a, qd2, CC * 9);
  k_quant_f32 <<<blks(CC * 9),  256, 0, stream>>>(c3dw_w, c3dw_a, qd3, CC * 9);
  k_quant_bf16<<<blks(CC * CC), 256, 0, stream>>>(c1pw_w, c1pw_a, qp1, CC * CC);
  k_quant_bf16<<<blks(CC * CC), 256, 0, stream>>>(c2pw_w, c2pw_a, qp2, CC * CC);
  k_quant_bf16<<<blks(QN * CC), 256, 0, stream>>>(c3pw_w, c3pw_a, qp3, QN * CC);
  k_quant_bf16<<<blks(CC * CC), 256, 0, stream>>>(fc1_w,  fc1_a,  qf1, CC * CC);
  k_quant_bf16<<<blks(CC * CC), 256, 0, stream>>>(fc2_w,  fc2_a,  qf2, CC * CC);

  // ---- 2) input prep ----
  k_prep<<<blks(NX), 256, 0, stream>>>(x, aF_a, xb, fq);
  k_padzero<<<blks(BN * CC * (HWP - HWN)), 256, 0, stream>>>(fq);

  // ---- 3) stage 1: dw5x5 -> pw (WMMA) -> GN -> PReLU ----
  k_dw<5, 2><<<blks(NX), 256, 0, stream>>>(xb, qd1, c1dw_b, d1);
  k_pw<<<dim3(4, HWN / 16, BN), 32, 0, stream>>>(d1, qp1, c1pw_b, y, CC, 1.0f);
  k_gnstats<<<BN * 32, 256, 0, stream>>>(y, st);
  k_gnapply<<<blks(NX), 256, 0, stream>>>(y, st, gn1_w, gn1_b, pr_amp, h1);

  // ---- 4) stage 2: dw3x3 -> pw (WMMA) -> GN -> PReLU ----
  k_dw<3, 1><<<blks(NX), 256, 0, stream>>>(h1, qd2, c2dw_b, d2);
  k_pw<<<dim3(4, HWN / 16, BN), 32, 0, stream>>>(d2, qp2, c2pw_b, y, CC, 1.0f);
  k_gnstats<<<BN * 32, 256, 0, stream>>>(y, st);
  k_gnapply<<<blks(NX), 256, 0, stream>>>(y, st, gn2_w, gn2_b, pr_amp, h2);

  // ---- 5) stage 3: dw3x3 -> pw (WMMA, Cout=300 padded to 320, *TAU) ----
  k_dw<3, 1><<<blks(NX), 256, 0, stream>>>(h2, qd3, c3dw_b, d3);
  k_pw<<<dim3(QW / 64, HWN / 16, BN), 32, 0, stream>>>(d3, qp3, c3pw_b,
                                                       amap, QN, TAU);

  // ---- 6) spatial softmax + LSQ(A) ----
  k_softmax<<<BN * QN, 256, 0, stream>>>(amap, aA_a, aqb);

  // ---- 7) weighted pooling BMM (WMMA): F_P = Aq x Fq ----
  k_bmm<<<dim3(CC / 64, QP / 16, BN), 32, 0, stream>>>(aqb, fq, fpv, fpb);

  // ---- 8) channel recalibration (WMMA linears) + gating -> d_out ----
  k_fc1<<<dim3(RQP / 64, CC / 16), 32, 0, stream>>>(qf1, fpb, fc1_b, pr_cr, hcrb);
  k_fc2<<<dim3(RQP / 64, CC / 16), 32, 0, stream>>>(qf2, hcrb, fc2_b, fpv,
                                                    aFC_a, aFP_a, outp);
}
